// PhaseMNIST_14121852469309
// MI455X (gfx1250) — matrix operand order repaired and checked by reading, not compiled, not run on hardware
//
#include <hip/hip_runtime.h>
#include <math.h>

// ---------------------------------------------------------------------------
// PhaseMNIST on MI455X (gfx1250): bf16 WMMA (16x16x32, f32 accum) for all
// GEMMs. Workgroup = 8 waves sharing one N-tile; the B panel (full-K x 16)
// is staged into LDS once per block via gfx1250 async global->LDS copies
// (ASYNCcnt / s_wait_asynccnt), then every wave streams only its A fragments
// from global while B fragments come from LDS (one 16-bit read per element).
// ---------------------------------------------------------------------------

typedef __bf16 bf16;
typedef __attribute__((ext_vector_type(16))) __bf16 v16bf;
typedef __attribute__((ext_vector_type(8)))  __bf16 v8bf;
typedef __attribute__((ext_vector_type(8)))  float  v8f;
typedef __attribute__((ext_vector_type(4)))  int    v4i;

#define NB   128
#define NP   256
#define NPIX 256
#define ND   512
#define NC   10
#define SCALE_ATTN 0.35355339059327373f   // 8/sqrt(512)
#define PI_F 3.14159265358979323846f

#if defined(__has_builtin)
#  if __has_builtin(__builtin_amdgcn_global_load_async_to_lds_b128)
#    define HAVE_ASYNC_LDS 1
#  else
#    define HAVE_ASYNC_LDS 0
#  endif
#else
#  define HAVE_ASYNC_LDS 0
#endif

#define GAS __attribute__((address_space(1)))
#define LAS __attribute__((address_space(3)))

// ---------------- WMMA fragment helpers (wave32) ---------------------------
// A (16x32 bf16, row-major, stride lda):
//   lane l holds row M = l&15; K-elements: e<8 -> (l>=16?8:0)+e, e>=8 -> +16.
__device__ __forceinline__ v16bf load_a_frag(const bf16* A, int lda, int lane) {
  const bf16* p = A + (lane & 15) * lda + ((lane >> 4) << 3);
  v8bf lo = *(const v8bf*)(p);
  v8bf hi = *(const v8bf*)(p + 16);
  return __builtin_shufflevector(lo, hi, 0,1,2,3,4,5,6,7,8,9,10,11,12,13,14,15);
}
// B fragment out of an LDS-resident K x 16 panel (row stride 16 elements):
// lane l holds col N = l&15; K-elements: e<8 -> k0+(l>=16?8:0)+e, e>=8 -> +16.
__device__ __forceinline__ v16bf lds_b_frag(const bf16* lds, int k0, int lane) {
  const bf16* p = lds + (k0 + ((lane >> 4) << 3)) * 16 + (lane & 15);
  v16bf r;
  for (int e = 0; e < 16; ++e) r[e] = p[((e & 7) + ((e >> 3) << 4)) * 16];
  return r;
}
__device__ __forceinline__ v8f wmma_bf16(v16bf a, v16bf b, v8f c) {
  return __builtin_amdgcn_wmma_f32_16x16x32_bf16(false, a, false, b,
                                                 (short)0, c, false, false);
}
// C/D layout: VGPR r -> row m0 + r + 8*(lane>>4), col n0 + (lane&15)

// Stage a KROWS x 16 bf16 panel (global row stride ldb elements) into LDS
// (row stride 16 elements). 16-byte chunks; KROWS*2 chunks over 256 threads.
template <int KROWS>
__device__ __forceinline__ void stage_b_panel(const bf16* __restrict__ gsrc,
                                              int ldb, bf16* lds) {
  constexpr int CH = KROWS * 2;                 // multiple of 256
  for (int s = threadIdx.x; s < CH; s += 256) {
    int row = s >> 1, half = s & 1;
    const bf16* g = gsrc + (size_t)row * ldb + half * 8;
    bf16* l = lds + row * 16 + half * 8;
#if HAVE_ASYNC_LDS
    __builtin_amdgcn_global_load_async_to_lds_b128(
        (GAS v4i*)(void*)const_cast<bf16*>(g),
        (LAS v4i*)(void*)l, 0, 0);
#else
    *(v8bf*)l = *(const v8bf*)g;
#endif
  }
}
__device__ __forceinline__ void stage_fence() {
#if HAVE_ASYNC_LDS
#  if __has_builtin(__builtin_amdgcn_s_wait_asynccnt)
  __builtin_amdgcn_s_wait_asynccnt(0);
#  else
  asm volatile("s_wait_asynccnt 0" ::: "memory");
#  endif
#endif
  __syncthreads();
}

// ---------------- init / conversion kernels --------------------------------
__global__ void k_init(const float* qrot, const float* krot, const float* vrot,
                       float* tab, int* flag) {
  int d = blockIdx.x * blockDim.x + threadIdx.x;
  if (d == 0) flag[0] = 1;
  if (d < ND) {
    tab[0*ND + d] = cosf(qrot[d]);  tab[1*ND + d] = sinf(qrot[d]);
    tab[2*ND + d] = cosf(krot[d]);  tab[3*ND + d] = sinf(krot[d]);
    tab[4*ND + d] = cosf(vrot[d]);  tab[5*ND + d] = sinf(vrot[d]);
  }
}

__global__ void k_f2b(const float* __restrict__ src, bf16* __restrict__ dst) {
  size_t i = (size_t)blockIdx.x * blockDim.x + threadIdx.x;
  dst[i] = (bf16)src[i];
}

// Wp (D, PIX) -> Wpt bf16 (PIX, D): Wpt[k*D+n] = Wp[n*PIX+k]
__global__ void k_wpt(const float* __restrict__ Wp, bf16* __restrict__ wpt) {
  int i = blockIdx.x * blockDim.x + threadIdx.x;   // over PIX*D
  int k = i >> 9, n = i & (ND - 1);
  wpt[i] = (bf16)Wp[(size_t)n * NPIX + k];
}

// ---------------- projection + phase init ----------------------------------
// mag = tanh(x @ Wp^T + bp); state = mag * exp(i * p * freq(d) * pi)
// Block: 8 waves share N-tile tn; wave w handles M-tile tm = (bid>>5)*8 + w.
__global__ void k_proj(const bf16* __restrict__ xbf, const bf16* __restrict__ wpt,
                       const float* __restrict__ bp,
                       float* __restrict__ st_r, float* __restrict__ st_i) {
  __shared__ __align__(16) bf16 LB[NPIX * 16];           // 8 KB
  int bid  = blockIdx.x;
  int tn   = bid & 31;
  int tm   = ((bid >> 5) << 3) + (threadIdx.x >> 5);
  int lane = threadIdx.x & 31;
  stage_b_panel<NPIX>(wpt + (tn << 4), ND, LB);
  stage_fence();
  int m0 = tm << 4, n0 = tn << 4;
  const bf16* Abase = xbf + (size_t)m0 * NPIX;
  v8f acc = {};
  for (int k = 0; k < NPIX; k += 32) {
    v16bf a = load_a_frag(Abase + k, NPIX, lane);
    acc = wmma_bf16(a, lds_b_frag(LB, k, lane), acc);
  }
  int col  = n0 + (lane & 15);
  float freq = __powf(10000.0f, -(float)col * (1.0f / (float)ND));
  float bias = bp[col];
  for (int r = 0; r < 8; ++r) {
    int row = m0 + r + ((lane >> 4) << 3);         // row = b*P + p
    float mag = tanhf(acc[r] + bias);
    float phase = (float)(row & (NP - 1)) * freq * PI_F;
    float s, c; __sincosf(phase, &s, &c);
    size_t idx = (size_t)row * ND + col;
    st_r[idx] = mag * c;
    st_i[idx] = mag * s;
  }
}

// ---------------- per-step rotation ----------------------------------------
// q = s*e^{i qrot}, k = s*e^{i krot} (stored TRANSPOSED (B,D,P)), v = s*e^{i vrot}
__global__ void k_rotate(const float* __restrict__ sr, const float* __restrict__ si,
                         const float* __restrict__ tab,
                         bf16* __restrict__ qr, bf16* __restrict__ qi,
                         bf16* __restrict__ krt, bf16* __restrict__ kit,
                         bf16* __restrict__ vr, bf16* __restrict__ vi) {
  size_t i = (size_t)blockIdx.x * blockDim.x + threadIdx.x;  // over B*P*D
  int d = (int)(i & (ND - 1));
  size_t row = i >> 9;                 // b*P + p
  int p = (int)(row & (NP - 1));
  int b = (int)(row >> 8);
  float xr = sr[i], xi = si[i];
  float cq = tab[0*ND + d], sq = tab[1*ND + d];
  float ck = tab[2*ND + d], sk = tab[3*ND + d];
  float cv = tab[4*ND + d], sv = tab[5*ND + d];
  qr[i] = (bf16)(xr * cq - xi * sq);
  qi[i] = (bf16)(xr * sq + xi * cq);
  float kr_ = xr * ck - xi * sk;
  float ki_ = xr * sk + xi * ck;
  size_t ti = ((size_t)b * ND + d) * NP + p;
  krt[ti] = (bf16)kr_;
  kit[ti] = (bf16)ki_;
  vr[i] = (bf16)(xr * cv - xi * sv);
  vi[i] = (bf16)(xr * sv + xi * cv);
}

// ---------------- scores: real(q . conj(k)) = qr@kr^T + qi@ki^T ------------
// Block: batch b, N-tile tn shared by 8 waves; wave w -> M-tile tm.
__global__ void k_scores(const bf16* __restrict__ qr, const bf16* __restrict__ qi,
                         const bf16* __restrict__ krt, const bf16* __restrict__ kit,
                         float* __restrict__ scores) {
  __shared__ __align__(16) bf16 LBR[ND * 16];            // 16 KB
  __shared__ __align__(16) bf16 LBI[ND * 16];            // 16 KB
  int bid  = blockIdx.x;
  int b    = bid >> 5;                 // 32 blocks per batch
  int r5   = bid & 31;
  int tn   = r5 & 15;
  int tm   = ((r5 >> 4) << 3) + (threadIdx.x >> 5);
  int lane = threadIdx.x & 31;
  const bf16* Bk = krt + (size_t)b * ND * NP + (tn << 4);
  const bf16* Bi = kit + (size_t)b * ND * NP + (tn << 4);
  stage_b_panel<ND>(Bk, NP, LBR);
  stage_b_panel<ND>(Bi, NP, LBI);
  stage_fence();
  const bf16* Aq = qr + ((size_t)b * NP + (tm << 4)) * ND;
  const bf16* Ai = qi + ((size_t)b * NP + (tm << 4)) * ND;
  v8f acc = {};
  for (int k = 0; k < ND; k += 32) {
    acc = wmma_bf16(load_a_frag(Aq + k, ND, lane), lds_b_frag(LBR, k, lane), acc);
    acc = wmma_bf16(load_a_frag(Ai + k, ND, lane), lds_b_frag(LBI, k, lane), acc);
  }
  int qcol = (tn << 4) + (lane & 15);
  for (int r = 0; r < 8; ++r) {
    int prow = (tm << 4) + r + ((lane >> 4) << 3);
    scores[((size_t)b * NP + prow) * NP + qcol] = acc[r];
  }
}

// ---------------- softmax over last dim (rows of length 256) ---------------
__global__ void k_softmax(const float* __restrict__ scores, bf16* __restrict__ attn) {
  __shared__ float sm[NP];
  size_t row = blockIdx.x;
  int t = threadIdx.x;
  float v = scores[row * NP + t] * SCALE_ATTN;
  sm[t] = v; __syncthreads();
  for (int s = NP / 2; s > 0; s >>= 1) {
    if (t < s) sm[t] = fmaxf(sm[t], sm[t + s]);
    __syncthreads();
  }
  float mx = sm[0]; __syncthreads();
  float e = __expf(v - mx);
  sm[t] = e; __syncthreads();
  for (int s = NP / 2; s > 0; s >>= 1) {
    if (t < s) sm[t] += sm[t + s];
    __syncthreads();
  }
  float inv = 1.0f / sm[0];
  attn[row * NP + t] = (bf16)(e * inv);
}

// ---------------- attn_out = attn @ v; t = prev + attn_out (bf16) ----------
// Block: batch b, N-tile tn shared by 8 waves; wave w -> M-tile tm.
__global__ void k_attnout(const bf16* __restrict__ attn,
                          const bf16* __restrict__ vr, const bf16* __restrict__ vi,
                          const float* __restrict__ sr, const float* __restrict__ si,
                          bf16* __restrict__ tr, bf16* __restrict__ ti_) {
  __shared__ __align__(16) bf16 LBR[NP * 16];            // 8 KB
  __shared__ __align__(16) bf16 LBI[NP * 16];            // 8 KB
  int bid  = blockIdx.x;
  int b    = bid >> 6;                 // 64 blocks per batch
  int r6   = bid & 63;
  int tn   = r6 & 31;
  int tm   = ((r6 >> 5) << 3) + (threadIdx.x >> 5);
  int lane = threadIdx.x & 31;
  stage_b_panel<NP>(vr + (size_t)b * NP * ND + (tn << 4), ND, LBR);
  stage_b_panel<NP>(vi + (size_t)b * NP * ND + (tn << 4), ND, LBI);
  stage_fence();
  const bf16* A = attn + ((size_t)b * NP + (tm << 4)) * NP;
  v8f ar = {}, ai = {};
  for (int k = 0; k < NP; k += 32) {
    v16bf a = load_a_frag(A + k, NP, lane);
    ar = wmma_bf16(a, lds_b_frag(LBR, k, lane), ar);
    ai = wmma_bf16(a, lds_b_frag(LBI, k, lane), ai);
  }
  int col = (tn << 4) + (lane & 15);
  for (int r = 0; r < 8; ++r) {
    int gr = b * NP + (tm << 4) + r + ((lane >> 4) << 3);
    size_t idx = (size_t)gr * ND + col;
    tr[idx]  = (bf16)(sr[idx] + ar[r]);
    ti_[idx] = (bf16)(si[idx] + ai[r]);
  }
}

// ---------------- y = t @ ff (complex): 4 real WMMA chains -----------------
// Block: N-tile tn shared by 8 waves; wave w -> M-tile tm.
__global__ void k_ff(const bf16* __restrict__ tr, const bf16* __restrict__ ti_,
                     const bf16* __restrict__ fr, const bf16* __restrict__ fi,
                     float* __restrict__ yr, float* __restrict__ yi) {
  __shared__ __align__(16) bf16 LBR[ND * 16];            // 16 KB
  __shared__ __align__(16) bf16 LBI[ND * 16];            // 16 KB
  int bid  = blockIdx.x;
  int tn   = bid & 31;
  int tm   = ((bid >> 5) << 3) + (threadIdx.x >> 5);
  int lane = threadIdx.x & 31;
  stage_b_panel<ND>(fr + (tn << 4), ND, LBR);
  stage_b_panel<ND>(fi + (tn << 4), ND, LBI);
  stage_fence();
  const bf16* Ar = tr  + (size_t)(tm << 4) * ND;
  const bf16* Ai = ti_ + (size_t)(tm << 4) * ND;
  v8f rr = {}, ri = {}, ir = {}, ii = {};
  for (int k = 0; k < ND; k += 32) {
    __builtin_prefetch(Ar + (size_t)(k + 32), 0, 1);
    v16bf a_r = load_a_frag(Ar + k, ND, lane);
    v16bf a_i = load_a_frag(Ai + k, ND, lane);
    v16bf b_r = lds_b_frag(LBR, k, lane);
    v16bf b_i = lds_b_frag(LBI, k, lane);
    rr = wmma_bf16(a_r, b_r, rr);
    ii = wmma_bf16(a_i, b_i, ii);
    ri = wmma_bf16(a_r, b_i, ri);
    ir = wmma_bf16(a_i, b_r, ir);
  }
  int col = (tn << 4) + (lane & 15);
  for (int r = 0; r < 8; ++r) {
    int row = (tm << 4) + r + ((lane >> 4) << 3);
    size_t idx = (size_t)row * ND + col;
    yr[idx] = rr[r] - ii[r];
    yi[idx] = ri[r] + ir[r];
  }
}

// ---------------- complex_norm (per-row over D) + diff partials ------------
__global__ void k_norm(float* __restrict__ nr, float* __restrict__ ni,
                       const float* __restrict__ pr, const float* __restrict__ pi,
                       float* __restrict__ partA, float* __restrict__ partB) {
  __shared__ float sm[256];
  size_t row = blockIdx.x;
  int t = threadIdx.x;
  size_t base = row * ND;
  float xr[2], xi[2], m[2];
  float lsum = 0.0f, lsq = 0.0f;
  for (int e = 0; e < 2; ++e) {
    size_t idx = base + t + e * 256;
    xr[e] = nr[idx]; xi[e] = ni[idx];
    m[e] = sqrtf(xr[e] * xr[e] + xi[e] * xi[e]);
    lsum += m[e]; lsq += m[e] * m[e];
  }
  sm[t] = lsum; __syncthreads();
  for (int s = 128; s > 0; s >>= 1) { if (t < s) sm[t] += sm[t + s]; __syncthreads(); }
  float mean = sm[0] * (1.0f / (float)ND); __syncthreads();
  sm[t] = lsq; __syncthreads();
  for (int s = 128; s > 0; s >>= 1) { if (t < s) sm[t] += sm[t + s]; __syncthreads(); }
  float var = (sm[0] - (float)ND * mean * mean) * (1.0f / (float)(ND - 1));
  float stdv = sqrtf(fmaxf(var, 0.0f)); __syncthreads();
  float dacc = 0.0f, nacc = 0.0f;
  for (int e = 0; e < 2; ++e) {
    size_t idx = base + t + e * 256;
    float nm = (m[e] - mean) / (stdv + 1e-5f);
    float sc = tanhf(nm) / (m[e] + 1e-5f);
    float oyr = xr[e] * sc, oyi = xi[e] * sc;
    nr[idx] = oyr; ni[idx] = oyi;
    float dr = oyr - pr[idx], di = oyi - pi[idx];
    dacc += dr * dr + di * di;
    nacc += oyr * oyr + oyi * oyi;
  }
  sm[t] = dacc; __syncthreads();
  for (int s = 128; s > 0; s >>= 1) { if (t < s) sm[t] += sm[t + s]; __syncthreads(); }
  if (t == 0) partA[row] = sm[0];
  __syncthreads();
  sm[t] = nacc; __syncthreads();
  for (int s = 128; s > 0; s >>= 1) { if (t < s) sm[t] += sm[t + s]; __syncthreads(); }
  if (t == 0) partB[row] = sm[0];
}

// ---------------- state = where(active, new, prev) -------------------------
__global__ void k_select(float* __restrict__ sr, float* __restrict__ si,
                         const float* __restrict__ nr, const float* __restrict__ ni,
                         const int* __restrict__ flag) {
  if (flag[0] == 0) return;
  size_t i = ((size_t)blockIdx.x * blockDim.x + threadIdx.x) * 4;
  *(float4*)(sr + i) = *(const float4*)(nr + i);
  *(float4*)(si + i) = *(const float4*)(ni + i);
}

// ---------------- active &= (diff >= TOL), deterministic reduce ------------
__global__ void k_flagupd(const float* __restrict__ partA,
                          const float* __restrict__ partB, int* flag) {
  __shared__ float sa[256], sb[256];
  int t = threadIdx.x;
  float a = 0.0f, b = 0.0f;
  for (int i = t; i < NB * NP; i += 256) { a += partA[i]; b += partB[i]; }
  sa[t] = a; sb[t] = b; __syncthreads();
  for (int s = 128; s > 0; s >>= 1) {
    if (t < s) { sa[t] += sa[t + s]; sb[t] += sb[t + s]; }
    __syncthreads();
  }
  if (t == 0) {
    float diff = sqrtf(sa[0]) / (sqrtf(sb[0]) + 1e-8f);
    if (diff < 1e-3f) flag[0] = 0;
  }
}

// ---------------- mean pool over P + logits --------------------------------
__global__ void k_pool(const float* __restrict__ sr, const float* __restrict__ si,
                       float* __restrict__ pr, float* __restrict__ pi) {
  int i = blockIdx.x * blockDim.x + threadIdx.x;     // over B*D
  int b = i >> 9, d = i & (ND - 1);
  size_t base = (size_t)b * NP * ND + d;
  float ar = 0.0f, ai = 0.0f;
  for (int p = 0; p < NP; ++p) { ar += sr[base + (size_t)p * ND]; ai += si[base + (size_t)p * ND]; }
  pr[i] = ar * (1.0f / (float)NP);
  pi[i] = ai * (1.0f / (float)NP);
}

__global__ void k_logits(const float* __restrict__ pr, const float* __restrict__ pi,
                         const float* __restrict__ Wr, const float* __restrict__ br,
                         const float* __restrict__ Wi, const float* __restrict__ bi,
                         float* __restrict__ out) {
  int i = blockIdx.x * blockDim.x + threadIdx.x;
  if (i >= NB * NC) return;
  int b = i / NC, c = i % NC;
  float acc = br[c] + bi[c];
  const float* prb = pr + (size_t)b * ND;
  const float* pib = pi + (size_t)b * ND;
  const float* wr = Wr + (size_t)c * ND;
  const float* wi = Wi + (size_t)c * ND;
  for (int d = 0; d < ND; ++d) acc += prb[d] * wr[d] + pib[d] * wi[d];
  out[i] = acc;
}

// ---------------------------------------------------------------------------
extern "C" void kernel_launch(void* const* d_in, const int* in_sizes, int n_in,
                              void* d_out, int out_size, void* d_ws, size_t ws_size,
                              hipStream_t stream) {
  const float* x    = (const float*)d_in[0];
  const float* Wp   = (const float*)d_in[1];
  const float* bp   = (const float*)d_in[2];
  const float* qrot = (const float*)d_in[3];
  const float* krot = (const float*)d_in[4];
  const float* vrot = (const float*)d_in[5];
  const float* ffr  = (const float*)d_in[6];
  const float* ffi  = (const float*)d_in[7];
  const float* Wr   = (const float*)d_in[8];
  const float* br   = (const float*)d_in[9];
  const float* Wi   = (const float*)d_in[10];
  const float* bi   = (const float*)d_in[11];
  float* out = (float*)d_out;

  const size_t NSD = (size_t)NB * NP * ND;    // 16,777,216
  const size_t NSP = (size_t)NB * NP * NP;    //  8,388,608
  const size_t NSX = (size_t)NB * NP * NPIX;  //  8,388,608

  char* cur = (char*)d_ws;
  auto take = [&](size_t bytes) -> char* {
    char* p = cur; cur += (bytes + 255) & ~(size_t)255; return p;
  };
  float* st_r  = (float*)take(NSD * 4);
  float* st_i  = (float*)take(NSD * 4);
  float* new_r = (float*)take(NSD * 4);
  float* new_i = (float*)take(NSD * 4);
  bf16*  xbf   = (bf16*)take(NSX * 2);
  bf16*  wpt   = (bf16*)take((size_t)NPIX * ND * 2);
  bf16*  ffrb  = (bf16*)take((size_t)ND * ND * 2);
  bf16*  ffib  = (bf16*)take((size_t)ND * ND * 2);
  float* tab   = (float*)take(6 * ND * 4);
  bf16*  qrb   = (bf16*)take(NSD * 2);
  bf16*  qib   = (bf16*)take(NSD * 2);
  bf16*  krtb  = (bf16*)take(NSD * 2);
  bf16*  kitb  = (bf16*)take(NSD * 2);
  bf16*  vrb   = (bf16*)take(NSD * 2);
  bf16*  vib   = (bf16*)take(NSD * 2);
  bf16*  trb   = (bf16*)take(NSD * 2);
  bf16*  tib   = (bf16*)take(NSD * 2);
  float* scores = (float*)take(NSP * 4);
  bf16*  attnb  = (bf16*)take(NSP * 2);
  float* pooled_r = (float*)take((size_t)NB * ND * 4);
  float* pooled_i = (float*)take((size_t)NB * ND * 4);
  float* partA = (float*)take((size_t)NB * NP * 4);
  float* partB = (float*)take((size_t)NB * NP * 4);
  int*   flag  = (int*)take(256);

  // --- operand prep ---
  k_init<<<2, 256, 0, stream>>>(qrot, krot, vrot, tab, flag);
  k_f2b<<<(unsigned)(NSX / 256), 256, 0, stream>>>(x, xbf);
  k_wpt<<<(NPIX * ND) / 256, 256, 0, stream>>>(Wp, wpt);
  k_f2b<<<(ND * ND) / 256, 256, 0, stream>>>(ffr, ffrb);
  k_f2b<<<(ND * ND) / 256, 256, 0, stream>>>(ffi, ffib);

  // --- projection + phase init: 256 tm-groups x 32 tn tiles ---
  k_proj<<<8192, 256, 0, stream>>>(xbf, wpt, bp, st_r, st_i);

  // --- 4 fixed refinement steps ---
  for (int step = 0; step < 4; ++step) {
    k_rotate<<<(unsigned)(NSD / 256), 256, 0, stream>>>(st_r, st_i, tab,
                                                        qrb, qib, krtb, kitb, vrb, vib);
    k_scores<<<4096, 256, 0, stream>>>(qrb, qib, krtb, kitb, scores);
    k_softmax<<<NB * NP, 256, 0, stream>>>(scores, attnb);
    k_attnout<<<8192, 256, 0, stream>>>(attnb, vrb, vib, st_r, st_i, trb, tib);
    k_ff<<<8192, 256, 0, stream>>>(trb, tib, ffrb, ffib, new_r, new_i);
    k_norm<<<NB * NP, 256, 0, stream>>>(new_r, new_i, st_r, st_i, partA, partB);
    k_select<<<(unsigned)(NSD / 4 / 256), 256, 0, stream>>>(st_r, st_i, new_r, new_i, flag);
    k_flagupd<<<1, 256, 0, stream>>>(partA, partB, flag);
  }

  // --- pool + logits ---
  k_pool<<<(NB * ND) / 256, 256, 0, stream>>>(st_r, st_i, pooled_r, pooled_i);
  k_logits<<<(NB * NC + 255) / 256, 256, 0, stream>>>(pooled_r, pooled_i,
                                                      Wr, br, Wi, bi, out);
  (void)in_sizes; (void)n_in; (void)out_size; (void)ws_size;
}
